// RoICenterNet_18073222381903
// MI455X (gfx1250) — compile-verified, matrix operand order checked by reading, not run on hardware
//
#include <hip/hip_runtime.h>
#include <hip/hip_bf16.h>

// ---------------------------------------------------------------------------
// RoICenterNet post-processing for MI455X (gfx1250, wave32).
//   B=16 images, N=2000 proposals, C=80 classes, MAX_DET=300.
// Phase A (score_kernel):  softmax over 81 logits/row using v_wmma row-sum
//                          reduction (16 rows x 32 k per WMMA, f32 accum),
//                          threshold + compact candidates via atomicAdd.
// Phase B (nms_kernel):    one workgroup per image; scores in LDS; 300
//                          iterations of block argmax + same-label IoU
//                          suppression (equivalent to the batched-NMS
//                          coordinate-offset trick in the reference).
// ---------------------------------------------------------------------------

typedef __attribute__((ext_vector_type(16))) _Float16 v16h;
typedef __attribute__((ext_vector_type(8)))  float    v8f;

#define BB      16
#define NN      2000
#define CP1     81          // C+1
#define NCLS    80
#define MAXDET  300
#define KMAX    8192        // per-image candidate cap (expect ~4.6k)
#define NEGF    (-1.0e9f)
#define NMS_T   1024

// --------------------------- Phase A ---------------------------------------
// 256 threads = 8 waves; each wave handles 16 rows of [B*N, 81] logits.
// grid = 32000 rows / 128 rows-per-block = 250 blocks.
__global__ __launch_bounds__(256)
void score_kernel(const float* __restrict__ logits,
                  const float* __restrict__ props,
                  int* __restrict__ cnt,
                  float* __restrict__ cScore,
                  float* __restrict__ cBox,
                  int* __restrict__ cLabel) {
    const int lane = threadIdx.x & 31;
    const int wid  = threadIdx.x >> 5;
    const int half = lane >> 4;      // which K-half of the A operand this lane feeds
    const int m    = lane & 15;      // matrix row within the 16-row tile
    const int row  = (blockIdx.x * 8 + wid) * 16 + m;   // < 32000 exactly

    const float* lg = logits + (size_t)row * CP1;
    __builtin_prefetch(lg, 0, 1);                        // global_prefetch_b8

    // B operand = all-ones (layout-independent), so D[m][n] = sum_k exp(row m).
    v16h ones;
#pragma unroll
    for (int j = 0; j < 16; ++j) ones[j] = (_Float16)1.0f;

    v8f acc = {};
#pragma unroll
    for (int c = 0; c < 3; ++c) {                        // 3 x K=32 covers 81 (pad 0)
        v16h a;
#pragma unroll
        for (int j = 0; j < 16; ++j) {
            // 16-bit A 16x32 layout: lane<16 -> K in {0..7,16..23},
            //                        lane>=16 -> K in {8..15,24..31}
            int k = 32 * c + j + 8 * half + ((j & 8) ? 8 : 0);
            float v = (k < CP1) ? __expf(lg[k]) : 0.0f;
            a[j] = (_Float16)v;
        }
        acc = __builtin_amdgcn_wmma_f32_16x16x32_f16(
            /*neg_a=*/false, a, /*neg_b=*/false, ones,
            /*c_mod=*/(short)0, acc, /*reuse_a=*/false, /*reuse_b=*/false);
    }

    // C/D layout: VGPR r holds M=r (lanes 0-15) / M=8+r (lanes 16-31), every
    // column identical.  Fetch row m's sum from lane ((m&8)<<1)|(m&7), elem m&7.
    float mine   = acc[lane & 7];
    int   src    = ((m & 8) << 1) | (m & 7);
    float rowSum = __shfl(mine, src, 32);
    float inv    = 1.0f / rowSum;

    const int img = row / NN;
    const float* pb = props + (size_t)row * 4;
    float px1 = pb[0], py1 = pb[1], px2 = pb[2], py2 = pb[3];
    bool areaOk = ((py2 - py1) * (px2 - px1)) > 0.1f;

    // Two lanes share each row: lane-half h takes classes k = 1+h, 3+h, ...
#pragma unroll 4
    for (int i = 0; i < 40; ++i) {
        int k = 1 + half + 2 * i;
        float s = __expf(lg[k]) * inv;
        if (s > 0.05f && areaOk) {
            int pos = atomicAdd(&cnt[img], 1);
            if (pos < KMAX) {
                size_t o = (size_t)img * KMAX + pos;
                cScore[o] = s;
                cLabel[o] = k;
                float* cb = cBox + o * 4;
                cb[0] = px1; cb[1] = py1; cb[2] = px2; cb[3] = py2;
            }
        }
    }
}

// --------------------------- Phase B ---------------------------------------
// One block per image. Mutable scores live in LDS (32 KB); candidate
// boxes/labels (~3 MB for all images) stay L2-resident (192 MB L2).
__global__ __launch_bounds__(NMS_T)
void nms_kernel(const int* __restrict__ cnt,
                const float* __restrict__ cScore,
                const float* __restrict__ cBox,
                const int* __restrict__ cLabel,
                float* __restrict__ outBoxes,   // [16,300,4]
                int*   __restrict__ outLabels,  // [16,300]
                float* __restrict__ outScores)  // [16,300]
{
    __shared__ float ls[KMAX];
    __shared__ float redV[32];
    __shared__ int   redI[32];
    __shared__ float bBox[4];
    __shared__ float bScore;
    __shared__ int   bLab;
    __shared__ int   bValid;

    const int img = blockIdx.x;
    const int tid = threadIdx.x;
    int K = cnt[img];
    if (K > KMAX) K = KMAX;

    const float* sc = cScore + (size_t)img * KMAX;
    const float* bx = cBox   + (size_t)img * KMAX * 4;
    const int*   lb = cLabel + (size_t)img * KMAX;

    for (int i = tid; i < KMAX; i += NMS_T) ls[i] = (i < K) ? sc[i] : NEGF;
    __syncthreads();

    float* oB = outBoxes  + img * MAXDET * 4;
    int*   oL = outLabels + img * MAXDET;
    float* oS = outScores + img * MAXDET;

    for (int t = 0; t < MAXDET; ++t) {
        // ---- block-wide argmax over ls[0..KMAX) ----
        float bv = 2.0f * NEGF;
        int   bi = 0;
        for (int i = tid; i < KMAX; i += NMS_T) {
            float v = ls[i];
            if (v > bv || (v == bv && i < bi)) { bv = v; bi = i; }
        }
#pragma unroll
        for (int off = 16; off; off >>= 1) {
            float ov = __shfl_xor(bv, off, 32);
            int   oi = __shfl_xor(bi, off, 32);
            if (ov > bv || (ov == bv && oi < bi)) { bv = ov; bi = oi; }
        }
        if ((tid & 31) == 0) { redV[tid >> 5] = bv; redI[tid >> 5] = bi; }
        __syncthreads();
        if (tid < 32) {
            bv = redV[tid]; bi = redI[tid];
#pragma unroll
            for (int off = 16; off; off >>= 1) {
                float ov = __shfl_xor(bv, off, 32);
                int   oi = __shfl_xor(bi, off, 32);
                if (ov > bv || (ov == bv && oi < bi)) { bv = ov; bi = oi; }
            }
            if (tid == 0) {
                int valid = (bv > NEGF * 0.5f) ? 1 : 0;
                float x1 = bx[bi * 4 + 0], y1 = bx[bi * 4 + 1];
                float x2 = bx[bi * 4 + 2], y2 = bx[bi * 4 + 3];
                int   lab = lb[bi];
                bBox[0] = x1; bBox[1] = y1; bBox[2] = x2; bBox[3] = y2;
                bScore = bv; bLab = lab; bValid = valid;
                if (valid) {
                    oB[t * 4 + 0] = x1; oB[t * 4 + 1] = y1;
                    oB[t * 4 + 2] = x2; oB[t * 4 + 3] = y2;
                    oL[t] = lab; oS[t] = bv;
                } else {
                    oB[t * 4 + 0] = 0.f; oB[t * 4 + 1] = 0.f;
                    oB[t * 4 + 2] = 0.f; oB[t * 4 + 3] = 0.f;
                    oL[t] = 0; oS[t] = 0.f;
                }
                ls[bi] = NEGF;                 // s.at[i].set(NEG)
            }
        }
        __syncthreads();

        // ---- suppression: same label AND IoU > 0.5 (offset-trick equiv.) ----
        if (bValid) {
            float x1 = bBox[0], y1 = bBox[1], x2 = bBox[2], y2 = bBox[3];
            int   lab = bLab;
            float ab  = (x2 - x1) * (y2 - y1);
            for (int i = tid; i < K; i += NMS_T) {
                if (lb[i] == lab && ls[i] > NEGF * 0.5f) {
                    float c0 = bx[i * 4 + 0], c1 = bx[i * 4 + 1];
                    float c2 = bx[i * 4 + 2], c3 = bx[i * 4 + 3];
                    float iw = fminf(x2, c2) - fmaxf(x1, c0);
                    float ih = fminf(y2, c3) - fmaxf(y1, c1);
                    float inter = fmaxf(iw, 0.f) * fmaxf(ih, 0.f);
                    float ar = (c3 - c1) * (c2 - c0);
                    float iou = inter / (ab + ar - inter + 1e-9f);
                    if (iou > 0.5f) ls[i] = NEGF;
                }
            }
        }
        __syncthreads();
    }
}

// --------------------------- launcher --------------------------------------
extern "C" void kernel_launch(void* const* d_in, const int* in_sizes, int n_in,
                              void* d_out, int out_size, void* d_ws, size_t ws_size,
                              hipStream_t stream) {
    const float* label_pre = (const float*)d_in[0];
    // d_in[1] (bbox_pre) is dead code in the reference — unused.
    const float* proposals = (const float*)d_in[2];

    // workspace layout
    int*   cnt    = (int*)d_ws;                                  // 16 ints
    float* cScore = (float*)((char*)d_ws + 64);                  // 16*KMAX f32
    float* cBox   = cScore + (size_t)BB * KMAX;                  // 16*KMAX*4 f32
    int*   cLabel = (int*)(cBox + (size_t)BB * KMAX * 4);        // 16*KMAX i32

    hipMemsetAsync(cnt, 0, 64, stream);

    // output layout: boxes [16,300,4] f32 | labels [16,300] i32 | scores [16,300] f32
    float* outBoxes  = (float*)d_out;
    int*   outLabels = (int*)(outBoxes + BB * MAXDET * 4);
    float* outScores = (float*)(outLabels + BB * MAXDET);

    score_kernel<<<250, 256, 0, stream>>>(label_pre, proposals,
                                          cnt, cScore, cBox, cLabel);
    nms_kernel<<<BB, NMS_T, 0, stream>>>(cnt, cScore, cBox, cLabel,
                                         outBoxes, outLabels, outScores);
}
